// SwarmByteRingModel_41729902248603
// MI455X (gfx1250) — matrix operand order repaired and checked by reading, not compiled, not run on hardware
//
#include <hip/hip_runtime.h>
#include <hip/hip_bf16.h>

// ---------------------------------------------------------------------------
// SwarmByteRingModel on MI455X (gfx1250)
//
// Recurrence serial in t (128) and being n (ring RAW) -> latency-bound.
// One persistent workgroup (1024 threads = 32 wave32) runs the whole scan on
// one WGP, using the 320KB CDNA5 LDS: proc_W resident as bf16 (128KB) +
// inp/s0/s1 activation buffers. Per-step [64x256]x[256x256] GEMM uses
// v_wmma_f32_16x16x32_bf16. Each wave owns two C tiles sharing the same n0:
// one B fragment feeds two interleaved independent accumulators, hiding the
// WMMA->WMMA hazard slots. Ring (16MB) lives in global ws (fits in 192MB L2).
// ---------------------------------------------------------------------------

#define B_  64
#define T_  128
#define M_  256
#define D_  256
#define N_  4
#define KW  5        // 2K+1 window

typedef __attribute__((ext_vector_type(16))) __bf16 v16bf;
typedef __attribute__((ext_vector_type(8)))  __bf16 v8bf;
typedef __attribute__((ext_vector_type(8)))  float  v8f;

// LDS layout (bytes). Total 303,104 <= 320KB WGP LDS.
#define OFF_W     0         // proc_W bf16 [256][256]          131072
#define OFF_INP   131072    // inp  f32 [64][256]               65536
#define OFF_S0    196608    // s0   bf16 [64][256] (A operand)  32768
#define OFF_S1    229376    // s1   f32 [64][256]               65536
#define OFF_WW    294912    // w_s  f32 [64][5]                  1280
#define OFF_IDX   296192    // idx_s i32 [64][5]                 1280
#define OFF_BASE  297472    // base_s i32 [64]                    256
#define OFF_P     297728    // p_s f32 [64]                       256
#define OFF_PTR   297984    // ptr_s f32 [4][64]                 1024
#define OFF_OACC  299008    // out accumulator f32 [64][8]       2048
#define OFF_RED   301056    // jump-gate partials f32 [64][8]    2048
#define LDS_BYTES 303104

__device__ __forceinline__ __bf16 f2bf(float f) {
    unsigned u = __builtin_bit_cast(unsigned, f);
    unsigned r = (u + 0x7FFFu + ((u >> 16) & 1u)) >> 16;   // RNE
    unsigned short h = (unsigned short)r;
    return __builtin_bit_cast(__bf16, h);
}

__device__ __forceinline__ float mod256(float v) {
    return v - 256.0f * floorf(v * (1.0f / 256.0f));
}

__global__ void swarm_init_state(float* __restrict__ st, long n) {
    long i = (long)blockIdx.x * blockDim.x + threadIdx.x;
    long stride = (long)gridDim.x * blockDim.x;
    for (; i < n; i += stride) st[i] = 0.0f;
}

__global__ __launch_bounds__(1024, 1)
void swarm_persistent_kernel(const float* __restrict__ x,
                             const float* __restrict__ in_W,
                             const float* __restrict__ in_b,
                             const float* __restrict__ out_W,
                             const float* __restrict__ out_b,
                             const float* __restrict__ proc_W,
                             const float* __restrict__ proc_b,
                             const float* __restrict__ dest,
                             const float* __restrict__ jg_W,
                             const float* __restrict__ jg_b,
                             const float* __restrict__ ctx_strength,
                             const float* __restrict__ phase,
                             const float* __restrict__ ptr_init,
                             float* __restrict__ out,
                             float* __restrict__ ring,   // [B,M,D]
                             float* __restrict__ hid)    // [N,B,D]
{
    extern __shared__ char smem[];
    __bf16* Wl     = (__bf16*)(smem + OFF_W);
    float*  inp    = (float*) (smem + OFF_INP);
    __bf16* s0     = (__bf16*)(smem + OFF_S0);
    float*  s1     = (float*) (smem + OFF_S1);
    float*  w_s    = (float*) (smem + OFF_WW);
    int*    idx_s  = (int*)   (smem + OFF_IDX);
    int*    base_s = (int*)   (smem + OFF_BASE);
    float*  p_s    = (float*) (smem + OFF_P);
    float*  ptr_s  = (float*) (smem + OFF_PTR);
    float*  oacc   = (float*) (smem + OFF_OACC);
    float*  red    = (float*) (smem + OFF_RED);

    const int tid = threadIdx.x;

    // One-time: weights -> LDS as bf16; pointer state -> LDS.
    for (int i = tid; i < D_ * D_; i += 1024) Wl[i] = f2bf(proc_W[i]);
    if (tid < N_ * B_) ptr_s[tid] = ptr_init[tid];
    __syncthreads();

    for (int t = 0; t < T_; ++t) {
        // ---- Phase A: shared input projection  inp = x_t @ in_W.T + in_b
        for (int i = tid; i < B_ * D_; i += 1024) {
            int b = i >> 8, d = i & 255;
            float acc = in_b[d];
            const float* xp = x + (b * T_ + t) * 8;
            const float* wp = in_W + d * 8;
            #pragma unroll
            for (int k = 0; k < 8; ++k) acc += xp[k] * wp[k];
            inp[i] = acc;
        }
        if (tid < B_ * 8) oacc[tid] = 0.0f;
        __syncthreads();

        for (int n = 0; n < N_; ++n) {
            // ---- Phase B: window indices + gaussian softmax weights
            if (tid < B_) {
                float p = ptr_s[n * B_ + tid];
                int base = (int)floorf(p);
                base = base < 0 ? 0 : (base > M_ - 1 ? M_ - 1 : base);
                base_s[tid] = base;
                p_s[tid] = p;
                float e[KW]; float se = 0.0f;
                #pragma unroll
                for (int j = 0; j < KW; ++j) {
                    int idx = (base + j - 2 + M_) & (M_ - 1);
                    idx_s[tid * KW + j] = idx;
                    float dd = (float)idx - p + 128.0f;
                    dd = mod256(dd) - 128.0f;
                    float ev = __expf(-(dd * dd) * 0.125f);   // /TEMP=8
                    e[j] = ev; se += ev;
                }
                float inv = 1.0f / se;
                #pragma unroll
                for (int j = 0; j < KW; ++j) w_s[tid * KW + j] = e[j] * inv;
            }
            __syncthreads();

            // ---- Phase C: ctx gather + tanh  -> s0 (bf16, WMMA A operand)
            float sigc = 1.0f / (1.0f + __expf(-ctx_strength[n]));
            for (int i = tid; i < B_ * D_; i += 1024) {
                int b = i >> 8, d = i & 255;
                float ctx = 0.0f;
                #pragma unroll
                for (int j = 0; j < KW; ++j)
                    ctx += w_s[b * KW + j] * ring[(b * M_ + idx_s[b * KW + j]) * D_ + d];
                float ph = (d < 16) ? phase[n * 16 + d] : 0.0f;
                float comb = inp[i] + sigc * ctx + 0.1f * ph + hid[(n * B_ + b) * D_ + d];
                s0[i] = f2bf(tanhf(comb));
            }
            __syncthreads();

            // ---- Phase D: s1 = tanh(s0 @ W.T + b) via WMMA bf16 16x16x32.
            // Wave w owns tiles (w) and (w+32): same n0 -> shared B fragment,
            // two independent accumulators interleaved to hide WMMA hazards.
            {
                const int lane = tid & 31, wave = tid >> 5;
                const int lm = lane & 15, hh = lane >> 4;     // half-select
                const int n0 = (wave & 15) << 4;
                const int m0 = (wave >> 4) << 4;              // 0 or 16
                const int m1 = m0 + 32;
                v8f acc0 = {}, acc1 = {};
                #pragma unroll
                for (int kc = 0; kc < 8; ++kc) {
                    const int k0 = kc * 32;
                    // B(k,j) = W[j,k]: 32 contiguous bytes of weight row (n0+lm)
                    const __bf16* bp = Wl + (n0 + lm) * D_ + k0 + hh * 16;
                    v16bf bb = *(const v16bf*)bp;
                    // A: lanes 0-15 -> K {0..15}, lanes 16-31 -> K {16..31}
                    const __bf16* ap0 = s0 + (m0 + lm) * D_ + k0 + hh * 16;
                    v8bf a0lo = *(const v8bf*)ap0;
                    v8bf a0hi = *(const v8bf*)(ap0 + 8);
                    const __bf16* ap1 = s0 + (m1 + lm) * D_ + k0 + hh * 16;
                    v8bf a1lo = *(const v8bf*)ap1;
                    v8bf a1hi = *(const v8bf*)(ap1 + 8);
                    v16bf a0 = __builtin_shufflevector(a0lo, a0hi,
                                 0,1,2,3,4,5,6,7,8,9,10,11,12,13,14,15);
                    v16bf a1 = __builtin_shufflevector(a1lo, a1hi,
                                 0,1,2,3,4,5,6,7,8,9,10,11,12,13,14,15);
                    acc0 = __builtin_amdgcn_wmma_f32_16x16x32_bf16(
                               false, a0, false, bb, (short)0, acc0, false, false);
                    acc1 = __builtin_amdgcn_wmma_f32_16x16x32_bf16(
                               false, a1, false, bb, (short)0, acc1, false, false);
                }
                const float bias = proc_b[n0 + lm];
                #pragma unroll
                for (int r = 0; r < 8; ++r) {                 // C layout: lanes16-31 -> M=8+r
                    s1[(m0 + r + hh * 8) * D_ + (n0 + lm)] = tanhf(acc0[r] + bias);
                    s1[(m1 + r + hh * 8) * D_ + (n0 + lm)] = tanhf(acc1[r] + bias);
                }
            }
            __syncthreads();

            // ---- Phase E: ring scatter-add + hid update
            for (int i = tid; i < B_ * D_; i += 1024) {
                int b = i >> 8, d = i & 255;
                float v = s1[i];
                hid[(n * B_ + b) * D_ + d] = v;
                #pragma unroll
                for (int j = 0; j < KW; ++j)
                    ring[(b * M_ + idx_s[b * KW + j]) * D_ + d] += w_s[b * KW + j] * v;
            }
            // ---- Phase F1: jump-gate partial dots (512 threads, 32 el each)
            if (tid < B_ * 8) {
                int b = tid >> 3, seg = tid & 7;
                const float* jw = jg_W + n * D_ + seg * 32;
                const float* sp = s1 + b * D_ + seg * 32;
                float pa = 0.0f;
                #pragma unroll
                for (int d = 0; d < 32; ++d) pa += sp[d] * jw[d];
                red[tid] = pa;
            }
            __syncthreads();

            // ---- Phase F2: gate + pointer update (threads 0..63)
            if (tid < B_) {
                float jl = jg_b[n];
                #pragma unroll
                for (int seg = 0; seg < 8; ++seg) jl += red[tid * 8 + seg];
                float pr = 1.0f / (1.0f + __expf(-jl));
                float g  = (pr > 0.5f) ? 1.0f : 0.0f;         // forward = hard gate
                float dst  = dest[n * M_ + base_s[tid]];
                float walk = mod256(p_s[tid] + 1.0f);
                ptr_s[n * B_ + tid] = mod256(g * dst + (1.0f - g) * walk);
            }
            // ---- Phase G: output projection, accumulated over beings
            if (tid < B_ * 8) {
                int b = tid >> 3, c = tid & 7;
                float a = out_b[c];
                const float* ow = out_W + c * D_;
                const float* sp = s1 + b * D_;
                for (int d = 0; d < D_; ++d) a += sp[d] * ow[d];
                oacc[tid] += a;
            }
            __syncthreads();
        }

        // ---- Phase H: mean combiner -> y[b,t,:]
        if (tid < B_ * 8) {
            int b = tid >> 3, c = tid & 7;
            out[(b * T_ + t) * 8 + c] = oacc[tid] * 0.25f;
        }
        __syncthreads();
    }
}

extern "C" void kernel_launch(void* const* d_in, const int* in_sizes, int n_in,
                              void* d_out, int out_size, void* d_ws, size_t ws_size,
                              hipStream_t stream) {
    (void)in_sizes; (void)n_in; (void)out_size; (void)ws_size;
    const float* x      = (const float*)d_in[0];
    const float* in_W   = (const float*)d_in[1];
    const float* in_b   = (const float*)d_in[2];
    const float* out_W  = (const float*)d_in[3];
    const float* out_b  = (const float*)d_in[4];
    const float* proc_W = (const float*)d_in[5];
    const float* proc_b = (const float*)d_in[6];
    const float* dest   = (const float*)d_in[7];
    const float* jg_W   = (const float*)d_in[8];
    const float* jg_b   = (const float*)d_in[9];
    const float* ctx_s  = (const float*)d_in[10];
    const float* phase  = (const float*)d_in[11];
    const float* ptr0   = (const float*)d_in[12];
    float* out = (float*)d_out;

    // workspace: ring [B,M,D] f32 then hid [N,B,D] f32
    float* ring = (float*)d_ws;
    float* hid  = ring + (long)B_ * M_ * D_;
    const long state_elems = (long)B_ * M_ * D_ + (long)N_ * B_ * D_;

    // opt in to >64KB dynamic LDS (320KB per WGP on CDNA5)
    (void)hipFuncSetAttribute((const void*)swarm_persistent_kernel,
                              hipFuncAttributeMaxDynamicSharedMemorySize, LDS_BYTES);

    swarm_init_state<<<2048, 256, 0, stream>>>(ring, state_elems);
    swarm_persistent_kernel<<<1, 1024, LDS_BYTES, stream>>>(
        x, in_W, in_b, out_W, out_b, proc_W, proc_b, dest, jg_W, jg_b,
        ctx_s, phase, ptr0, out, ring, hid);
}
